// M_D_58213986730364
// MI455X (gfx1250) — compile-verified, matrix operand order checked by reading, not run on hardware
//
#include <hip/hip_runtime.h>
#include <hip/hip_bf16.h>

// ---------------------------------------------------------------------------
// Problem constants (from reference)
// ---------------------------------------------------------------------------
#define TT   64
#define NN   4096
#define ZZ   256
#define GG   1024          // 4*Z
#define FF   128
#define DTOT 409           // 300 + 74 + 35
#define KMAX 576           // largest padded K (modality 'l': 320 + 256)

typedef __attribute__((ext_vector_type(16))) __bf16 v16bf;
typedef __attribute__((ext_vector_type(8)))  float  v8f;

union Frag {
  v16bf  v;
  __bf16 e[16];
  uint4  q[2];
};

__device__ __forceinline__ float sigmoidf_(float x) { return 1.0f / (1.0f + __expf(-x)); }

// ---------------------------------------------------------------------------
// Weight packing: Wcat[g][k] = k<D ? W_ih[g][k] : (k<Dpad ? 0 : W_hh[g][k-Dpad])
// ---------------------------------------------------------------------------
__global__ void k_pack_wcat(const float* __restrict__ Wih, const float* __restrict__ Whh,
                            __bf16* __restrict__ out, int D, int Dpad, int K) {
  int idx = blockIdx.x * 256 + threadIdx.x;
  if (idx >= GG * K) return;
  int g = idx / K, k = idx - g * K;
  float v = 0.0f;
  if (k < D)          v = Wih[(size_t)g * D + k];
  else if (k >= Dpad) v = Whh[(size_t)g * ZZ + (k - Dpad)];
  out[idx] = (__bf16)v;
}

__global__ void k_cvt_bf16(const float* __restrict__ in, __bf16* __restrict__ out, int n) {
  int idx = blockIdx.x * 256 + threadIdx.x;
  if (idx < n) out[idx] = (__bf16)in[idx];
}

__global__ void k_bias_sum(const float* __restrict__ bih, const float* __restrict__ bhh,
                           float* __restrict__ out) {
  int idx = blockIdx.x * 256 + threadIdx.x;
  if (idx < GG) out[idx] = bih[idx] + bhh[idx];
}

// ---------------------------------------------------------------------------
// Slice copy: x (T,N,409) -> x_l (T,N,300), x_a (T,N,74), x_v (T,N,35)
// ---------------------------------------------------------------------------
__global__ void k_slice(const float* __restrict__ x, float* __restrict__ ol,
                        float* __restrict__ oa, float* __restrict__ ov) {
  size_t idx = (size_t)blockIdx.x * 256 + threadIdx.x;
  const size_t total = (size_t)TT * NN * DTOT;
  if (idx >= total) return;
  size_t row = idx / DTOT;
  int d = (int)(idx - row * DTOT);
  float v = x[idx];
  if (d < 300)      ol[row * 300 + d] = v;
  else if (d < 374) oa[row * 74 + (d - 300)] = v;
  else              ov[row * 35 + (d - 374)] = v;
}

// ---------------------------------------------------------------------------
// Persistent LSTM scan kernel.
// Grid: (N/16, 3).  Block: 256 threads = 8 waves (wave32).
// Block owns 16 batch rows for the whole 64-step scan.
// LDS A-panel sa[16][K] = [x_t | 0-pad | h]: x refreshed per step by a
// cooperative fill phase, h written by the elementwise phase. GEMM A-fragments
// are uniform ds_load_b128 pairs; B tiles stream from L2 (packed bf16 Wcat).
// ---------------------------------------------------------------------------
__global__ __launch_bounds__(256) void k_lstm_scan(
    const float*  __restrict__ x,
    const __bf16* __restrict__ wcat_l,
    const __bf16* __restrict__ wcat_a,
    const __bf16* __restrict__ wcat_v,
    const float*  __restrict__ bias_all,   // 3*1024 (b_ih + b_hh)
    __bf16*       __restrict__ hbf_all)    // 3*4096*256 (final h, bf16)
{
  __shared__ float sg[16 * GG];                    // 64 KB gate pre-activations
  __shared__ __align__(16) __bf16 sa[16 * KMAX];   // 18 KB A-panel [x | pad | h]
  __shared__ float sc[16 * ZZ];                    // 16 KB cell state

  const int tid  = threadIdx.x;
  const int lane = tid & 31;
  const int wave = tid >> 5;
  const int m    = blockIdx.y;
  const int rowBase = blockIdx.x * 16;

  const __bf16* wcat = (m == 0) ? wcat_l : ((m == 1) ? wcat_a : wcat_v);
  const int D    = (m == 0) ? 300 : ((m == 1) ? 74 : 35);
  const int Dpad = (m == 0) ? 320 : ((m == 1) ? 96 : 64);
  const int K    = Dpad + ZZ;
  const int xoff = (m == 0) ? 0 : ((m == 1) ? 300 : 374);
  const float* bias = bias_all + m * GG;
  __bf16* hbf = hbf_all + (size_t)m * NN * ZZ;

  // zero panel (covers pad columns + initial h = 0) and cell state
  for (int i = tid; i < 16 * KMAX; i += 256) sa[i] = (__bf16)0.0f;
  for (int i = tid; i < 16 * ZZ;   i += 256) sc[i] = 0.0f;
  __syncthreads();

  const int r15 = lane & 15;            // row within 16-row tile / col within 16-col tile
  const int khi = (lane >> 4) << 3;     // K sub-offset: 0 or 8
  const int rb  = (lane >> 4) << 3;     // C/D row offset: 0 or 8
  const int colBase = wave * 128;
  const int fillN = 16 * D;

  for (int t = 0; t < TT; ++t) {
    // ---- fill phase: x_t slice (16 x D fp32) -> bf16 panel, coalesced ----
    {
      const float* xbase = x + ((size_t)t * NN + rowBase) * DTOT + xoff;
      for (int i = tid; i < fillN; i += 256) {
        const int row = i / D;
        const int k   = i - row * D;
        sa[row * K + k] = (__bf16)xbase[(size_t)row * DTOT + k];
      }
    }
    __syncthreads();   // fill + previous h-writes visible before GEMM reads

    // ---- GEMM phase: gates(16x1024) = sa(16xK) @ Wcat^T ----
    v8f acc[8];
    #pragma unroll
    for (int i = 0; i < 8; ++i) {
      #pragma unroll
      for (int j = 0; j < 8; ++j) acc[i][j] = 0.0f;
    }

    for (int k0 = 0; k0 < K; k0 += 32) {
      const int kb = k0 + khi;
      Frag a;
      const __bf16* ar = &sa[r15 * K + kb];
      a.q[0] = *(const uint4*)(ar);
      a.q[1] = *(const uint4*)(ar + 16);
      #pragma unroll
      for (int tile = 0; tile < 8; ++tile) {
        const int c = colBase + tile * 16 + r15;
        const __bf16* wr = wcat + (size_t)c * K + kb;
        Frag b;
        b.q[0] = *(const uint4*)(wr);
        b.q[1] = *(const uint4*)(wr + 16);
        acc[tile] = __builtin_amdgcn_wmma_f32_16x16x32_bf16(
            false, a.v, false, b.v, (short)0, acc[tile], false, false);
      }
    }

    // stage gate pre-activations (+bias) to LDS
    #pragma unroll
    for (int tile = 0; tile < 8; ++tile) {
      const int c = colBase + tile * 16 + r15;
      const float bb = bias[c];
      #pragma unroll
      for (int v = 0; v < 8; ++v) sg[(v + rb) * GG + c] = acc[tile][v] + bb;
    }
    __syncthreads();   // sg visible; all GEMM panel-reads complete

    // ---- elementwise LSTM cell update (each thread: 1 row x 16 z) ----
    {
      const int row = tid >> 4;
      const int z0  = (tid & 15) << 4;
      #pragma unroll
      for (int j = 0; j < 16; ++j) {
        const int z = z0 + j;
        const float gi = sigmoidf_(sg[row * GG + z]);
        const float gf = sigmoidf_(sg[row * GG + ZZ + z]);
        const float gg = tanhf    (sg[row * GG + 2 * ZZ + z]);
        const float go = sigmoidf_(sg[row * GG + 3 * ZZ + z]);
        const float cv = gf * sc[row * ZZ + z] + gi * gg;
        sc[row * ZZ + z] = cv;
        sa[row * K + Dpad + z] = (__bf16)(go * tanhf(cv));  // h into panel tail
      }
    }
    // no barrier needed here: next fill writes only the (disjoint) x columns,
    // and every thread's GEMM reads completed before the barrier above.
  }

  __syncthreads();
  // flush final hidden state (bf16) for the downstream Wfc GEMM
  for (int i = tid; i < 16 * ZZ; i += 256)
    hbf[(size_t)(rowBase + (i >> 8)) * ZZ + (i & (ZZ - 1))] = sa[(i >> 8) * K + Dpad + (i & (ZZ - 1))];
}

// ---------------------------------------------------------------------------
// Generic WMMA GEMM: out(N x Nout) = A(N x K) @ W^T + bias, optional ReLU.
// Block: 256 threads = 8 waves; each wave: 16 rows x Nout cols (NT = Nout/16).
// Grid: N/128.  ABF16: A is bf16 (else fp32 converted on the fly).
// ---------------------------------------------------------------------------
template <int NT, bool ABF16>
__global__ __launch_bounds__(256) void k_gemm(
    const void* __restrict__ Ain, int K,
    const __bf16* __restrict__ W, const float* __restrict__ bias,
    float* __restrict__ out, int ldo, int ooff, int relu)
{
  const int tid  = threadIdx.x;
  const int lane = tid & 31;
  const int wave = tid >> 5;
  const int r15  = lane & 15;
  const int khi  = (lane >> 4) << 3;
  const int rb   = (lane >> 4) << 3;
  const int tileRowBase = blockIdx.x * 128 + wave * 16;
  const int r = tileRowBase + r15;

  v8f acc[NT];
  #pragma unroll
  for (int i = 0; i < NT; ++i) {
    #pragma unroll
    for (int j = 0; j < 8; ++j) acc[i][j] = 0.0f;
  }

  for (int k0 = 0; k0 < K; k0 += 32) {
    Frag a;
    const int kb = k0 + khi;
    if (ABF16) {
      const __bf16* ar = (const __bf16*)Ain + (size_t)r * K + kb;
      a.q[0] = *(const uint4*)(ar);
      a.q[1] = *(const uint4*)(ar + 16);
    } else {
      const float* ar = (const float*)Ain + (size_t)r * K + kb;
      #pragma unroll
      for (int j = 0; j < 8; ++j) a.e[j]     = (__bf16)ar[j];
      #pragma unroll
      for (int j = 0; j < 8; ++j) a.e[8 + j] = (__bf16)ar[16 + j];
    }
    #pragma unroll
    for (int tile = 0; tile < NT; ++tile) {
      const int c = tile * 16 + r15;
      const __bf16* wr = W + (size_t)c * K + kb;
      Frag b;
      b.q[0] = *(const uint4*)(wr);
      b.q[1] = *(const uint4*)(wr + 16);
      acc[tile] = __builtin_amdgcn_wmma_f32_16x16x32_bf16(
          false, a.v, false, b.v, (short)0, acc[tile], false, false);
    }
  }

  #pragma unroll
  for (int tile = 0; tile < NT; ++tile) {
    const int c = tile * 16 + r15;
    const float bb = bias[c];
    #pragma unroll
    for (int v = 0; v < 8; ++v) {
      float val = acc[tile][v] + bb;
      if (relu) val = fmaxf(val, 0.0f);
      out[(size_t)(tileRowBase + v + rb) * ldo + ooff + c] = val;
    }
  }
}

// ---------------------------------------------------------------------------
// Final projection: y[n] = dot(fs[n, 0:384], Wy) + by
// ---------------------------------------------------------------------------
__global__ void k_y(const float* __restrict__ fs, const float* __restrict__ Wy,
                    const float* __restrict__ by, float* __restrict__ out) {
  int n = blockIdx.x * 256 + threadIdx.x;
  if (n >= NN) return;
  float acc = by[0];
  const float* f = fs + (size_t)n * (3 * FF);
  for (int k = 0; k < 3 * FF; ++k) acc += f[k] * Wy[k];
  out[n] = acc;
}

// ---------------------------------------------------------------------------
// Host launcher
// ---------------------------------------------------------------------------
extern "C" void kernel_launch(void* const* d_in, const int* in_sizes, int n_in,
                              void* d_out, int out_size, void* d_ws, size_t ws_size,
                              hipStream_t stream) {
  (void)in_sizes; (void)n_in; (void)out_size; (void)ws_size;

  const float* x = (const float*)d_in[0];
  const float* Wih[3] = {(const float*)d_in[1],  (const float*)d_in[11], (const float*)d_in[21]};
  const float* Whh[3] = {(const float*)d_in[2],  (const float*)d_in[12], (const float*)d_in[22]};
  const float* bih[3] = {(const float*)d_in[3],  (const float*)d_in[13], (const float*)d_in[23]};
  const float* bhh[3] = {(const float*)d_in[4],  (const float*)d_in[14], (const float*)d_in[24]};
  const float* Wfc[3] = {(const float*)d_in[5],  (const float*)d_in[15], (const float*)d_in[25]};
  const float* bfc[3] = {(const float*)d_in[6],  (const float*)d_in[16], (const float*)d_in[26]};
  const float* Wh1[3] = {(const float*)d_in[7],  (const float*)d_in[17], (const float*)d_in[27]};
  const float* bh1[3] = {(const float*)d_in[8],  (const float*)d_in[18], (const float*)d_in[28]};
  const float* Wh2[3] = {(const float*)d_in[9],  (const float*)d_in[19], (const float*)d_in[29]};
  const float* bh2[3] = {(const float*)d_in[10], (const float*)d_in[20], (const float*)d_in[30]};
  const float* Wy = (const float*)d_in[31];
  const float* by = (const float*)d_in[32];

  float* out   = (float*)d_out;
  float* out_l = out;
  float* out_a = out + (size_t)TT * NN * 300;
  float* out_v = out_a + (size_t)TT * NN * 74;
  float* out_y = out + (size_t)TT * NN * DTOT;

  // workspace carve-out (256B aligned)
  char* p = (char*)d_ws;
  auto alloc = [&](size_t bytes) -> char* {
    char* r = p; p += (bytes + 255) & ~(size_t)255; return r;
  };
  const int Km[3] = {576, 352, 320};
  const int Dm[3] = {300, 74, 35};
  const int Dp[3] = {320, 96, 64};

  __bf16* wcat[3];
  for (int m = 0; m < 3; ++m) wcat[m] = (__bf16*)alloc((size_t)GG * Km[m] * 2);
  float*  biasAll = (float*)alloc((size_t)3 * GG * 4);
  __bf16* wfcB    = (__bf16*)alloc((size_t)3 * ZZ * ZZ * 2);
  __bf16* wh1B    = (__bf16*)alloc((size_t)3 * FF * ZZ * 2);
  __bf16* wh2B    = (__bf16*)alloc((size_t)3 * FF * FF * 2);
  __bf16* hbfAll  = (__bf16*)alloc((size_t)3 * NN * ZZ * 2);
  float*  zAll    = (float*)alloc((size_t)3 * NN * ZZ * 4);
  float*  f1All   = (float*)alloc((size_t)3 * NN * FF * 4);
  float*  fsBuf   = (float*)alloc((size_t)NN * 3 * FF * 4);

  // 1. weight prep
  for (int m = 0; m < 3; ++m) {
    k_pack_wcat<<<(GG * Km[m] + 255) / 256, 256, 0, stream>>>(Wih[m], Whh[m], wcat[m], Dm[m], Dp[m], Km[m]);
    k_cvt_bf16<<<(ZZ * ZZ + 255) / 256, 256, 0, stream>>>(Wfc[m], wfcB + (size_t)m * ZZ * ZZ, ZZ * ZZ);
    k_cvt_bf16<<<(FF * ZZ + 255) / 256, 256, 0, stream>>>(Wh1[m], wh1B + (size_t)m * FF * ZZ, FF * ZZ);
    k_cvt_bf16<<<(FF * FF + 255) / 256, 256, 0, stream>>>(Wh2[m], wh2B + (size_t)m * FF * FF, FF * FF);
    k_bias_sum<<<4, 256, 0, stream>>>(bih[m], bhh[m], biasAll + m * GG);
  }

  // 2. slice copies (x_l, x_a, x_v outputs)
  {
    const size_t total = (size_t)TT * NN * DTOT;
    k_slice<<<(unsigned)((total + 255) / 256), 256, 0, stream>>>(x, out_l, out_a, out_v);
  }

  // 3. persistent LSTM scan (all 64 steps, all 3 modalities)
  {
    dim3 grid(NN / 16, 3);
    k_lstm_scan<<<grid, 256, 0, stream>>>(x, wcat[0], wcat[1], wcat[2], biasAll, hbfAll);
  }

  // 4. z = h @ Wfc^T + bfc  (bf16 A, Nout=256, no relu)
  for (int m = 0; m < 3; ++m)
    k_gemm<16, true><<<NN / 128, 256, 0, stream>>>(
        hbfAll + (size_t)m * NN * ZZ, ZZ, wfcB + (size_t)m * ZZ * ZZ, bfc[m],
        zAll + (size_t)m * NN * ZZ, ZZ, 0, 0);

  // 5. f1 = relu(z @ Wh1^T + b1)  (fp32 A, Nout=128)
  for (int m = 0; m < 3; ++m)
    k_gemm<8, false><<<NN / 128, 256, 0, stream>>>(
        zAll + (size_t)m * NN * ZZ, ZZ, wh1B + (size_t)m * FF * ZZ, bh1[m],
        f1All + (size_t)m * NN * FF, FF, 0, 1);

  // 6. f2 = relu(f1 @ Wh2^T + b2), written directly into fs concat slot
  for (int m = 0; m < 3; ++m)
    k_gemm<8, false><<<NN / 128, 256, 0, stream>>>(
        f1All + (size_t)m * NN * FF, FF, wh2B + (size_t)m * FF * FF, bh2[m],
        fsBuf, 3 * FF, m * FF, 1);

  // 7. y_hat
  k_y<<<NN / 256, 256, 0, stream>>>(fsBuf, Wy, by, out_y);
}
